// SwinTransformerBlock_20366734917789
// MI455X (gfx1250) — matrix operand order repaired and checked by reading, not compiled, not run on hardware
//
#include <hip/hip_runtime.h>

// ---------------- problem constants ----------------
constexpr int BB   = 2;
constexpr int SS   = 14, HH = 28, WWD = 28;
constexpr int CC   = 128;
constexpr int NHD  = 8;            // heads
constexpr int HDM  = 16;           // head dim
constexpr int WSZ  = 7, SHF = 3;
constexpr int NT   = 343;          // tokens per window
constexpr int NPAD = 352;          // 22 * 16
constexpr int NWIN = 64;           // total windows (B * 32)
constexpr int HID  = 512;
constexpr int TOK  = BB * SS * HH * WWD;   // 21952
constexpr float EPSV  = 1e-5f;
constexpr float SCALE = 0.25f;             // (C/NH)^-0.5
constexpr int MT_WIN = NPAD / 16;          // 22 M-tiles per window
constexpr int MT_TOT = NWIN * MT_WIN;      // 1408

// ---------------- WMMA types / helpers ----------------
typedef __attribute__((ext_vector_type(16))) __bf16 bf16x16;
typedef __attribute__((ext_vector_type(8)))  float  f32x8;

static __device__ __forceinline__ f32x8 wmma_bf16(bf16x16 a, bf16x16 b, f32x8 c) {
  return __builtin_amdgcn_wmma_f32_16x16x32_bf16(false, a, false, b, (short)0, c,
                                                 false, false);
}

static __device__ __forceinline__ unsigned short f2bf(float x) {
  unsigned int u = __float_as_uint(x);
  unsigned int r = u + 0x7FFFu + ((u >> 16) & 1u);   // round-to-nearest-even
  return (unsigned short)(r >> 16);
}

union FragU { bf16x16 v; unsigned int d[8]; };

// A fragment: 16x32 bf16 tile, row-major source, ld in elements.
// Per ISA 7.12.2: lane m=lane&15, VGPR r holds K = 16*(r>>2) + 8*(lane>>4) + 2*(r&3) (+e)
static __device__ __forceinline__ bf16x16 load_a_frag(const unsigned short* base, int ld, int lane) {
  FragU f;
  const unsigned short* rp = base + (size_t)(lane & 15) * ld;
  int kb = (lane >> 4) << 3;
#pragma unroll
  for (int r = 0; r < 8; ++r) {
    int K = ((r >> 2) << 4) + kb + ((r & 3) << 1);
    f.d[r] = *(const unsigned int*)(rp + K);
  }
  return f.v;
}

// A fragment where only K<16 is valid (zero-pad K=16..31); source ld elements.
static __device__ __forceinline__ bf16x16 load_a_frag_k16(const unsigned short* base, int ld, int lane) {
  FragU f;
  const unsigned short* rp = base + (size_t)(lane & 15) * ld;
  int kb = (lane >> 4) << 3;
#pragma unroll
  for (int r = 0; r < 8; ++r) {
    if (r < 4) f.d[r] = *(const unsigned int*)(rp + kb + ((r & 3) << 1));
    else       f.d[r] = 0u;
  }
  return f.v;
}

// B fragment: 32x16 bf16 (KxN) sourced from Bt stored [N][K] row-major.
// Per ISA: lane n=lane&15, VGPR r holds K = 16*(r>>2) + 8*(lane>>4) + 2*(r&3) (+e)
static __device__ __forceinline__ bf16x16 load_b_frag(const unsigned short* baseT, int ld, int lane) {
  FragU f;
  const unsigned short* rp = baseT + (size_t)(lane & 15) * ld + ((lane >> 4) << 3);
#pragma unroll
  for (int r = 0; r < 8; ++r) f.d[r] = *(const unsigned int*)(rp + ((r >> 2) << 4) + ((r & 3) << 1));
  return f.v;
}

// B fragment with only K<16 valid (VGPRs 4..7 are K>=16 -> zero).
static __device__ __forceinline__ bf16x16 load_b_frag_k16(const unsigned short* baseT, int ld, int lane) {
  FragU f;
  const unsigned short* rp = baseT + (size_t)(lane & 15) * ld + ((lane >> 4) << 3);
#pragma unroll
  for (int r = 0; r < 8; ++r) f.d[r] = (r < 4) ? *(const unsigned int*)(rp + (r << 1)) : 0u;
  return f.v;
}

static __device__ __forceinline__ f32x8 zero8() {
  f32x8 c = {0.f, 0.f, 0.f, 0.f, 0.f, 0.f, 0.f, 0.f};
  return c;
}

// ---------------- kernel 0: weight f32 -> bf16, transposed to [N][K] ----------------
__global__ __launch_bounds__(256) void transpose_bf16_kernel(const float* __restrict__ w,
                                                             unsigned short* __restrict__ out,
                                                             int K, int Nn) {
  int idx = blockIdx.x * 256 + threadIdx.x;
  if (idx >= K * Nn) return;
  int n = idx / K, k = idx - n * K;
  out[idx] = f2bf(w[(size_t)k * Nn + n]);
}

// ---------------- kernel 1: LN1 + roll(-3) + window partition -> bf16 ----------------
__global__ __launch_bounds__(256) void ln1_window_kernel(const float* __restrict__ x,
                                                         const float* __restrict__ g1,
                                                         const float* __restrict__ b1,
                                                         unsigned short* __restrict__ hwin) {
  int wid = threadIdx.x >> 5, lane = threadIdx.x & 31;
  int t = blockIdx.x * 8 + wid;            // padded token: win*NPAD + n
  if (t >= NWIN * NPAD) return;
  int win = t / NPAD, n = t - win * NPAD;
  unsigned short* dst = hwin + (size_t)t * CC;
  if (n >= NT) {                           // zero pad rows
#pragma unroll
    for (int r = 0; r < 4; ++r) dst[lane + 32 * r] = 0;
    return;
  }
  int b = win >> 5, wr = win & 31;
  int wsI = wr >> 4, whI = (wr >> 2) & 3, wwI = wr & 3;
  int i = n / 49, rem = n - i * 49, j = rem / 7, k = rem - j * 7;
  int ss = (wsI * WSZ + i + SHF) % SS;
  int hh = (whI * WSZ + j + SHF) % HH;
  int wc = (wwI * WSZ + k + SHF) % WWD;
  const float* row = x + (size_t)(((b * SS + ss) * HH + hh) * WWD + wc) * CC;
  float v[4]; float s = 0.f;
#pragma unroll
  for (int r = 0; r < 4; ++r) { v[r] = row[lane + 32 * r]; s += v[r]; }
#pragma unroll
  for (int o = 16; o >= 1; o >>= 1) s += __shfl_xor(s, o, 32);
  float mu = s * (1.f / 128.f);
  float vs = 0.f;
#pragma unroll
  for (int r = 0; r < 4; ++r) { v[r] -= mu; vs += v[r] * v[r]; }
#pragma unroll
  for (int o = 16; o >= 1; o >>= 1) vs += __shfl_xor(vs, o, 32);
  float rstd = rsqrtf(vs * (1.f / 128.f) + EPSV);
#pragma unroll
  for (int r = 0; r < 4; ++r) {
    int c = lane + 32 * r;
    dst[c] = f2bf(v[r] * rstd * g1[c] + b1[c]);
  }
}

// ---------------- kernel 2: QKV GEMM (16x64 strip per wave), route q/k/vT ----------------
__global__ __launch_bounds__(256) void qkv_kernel(const unsigned short* __restrict__ hwin,
                                                  const unsigned short* __restrict__ qkv_wT,
                                                  const float* __restrict__ qkv_b,
                                                  unsigned short* __restrict__ qs,
                                                  unsigned short* __restrict__ ks,
                                                  unsigned short* __restrict__ vT) {
  int wid = threadIdx.x >> 5, lane = threadIdx.x & 31;
  int strip = blockIdx.x * 8 + wid;        // MT_TOT * 6 strips (24 N-tiles / 4)
  if (strip >= MT_TOT * 6) return;
  int tm = strip / 6, tg = strip - tm * 6;
  int tn0 = tg * 4;
  const unsigned short* abase = hwin + (size_t)tm * 16 * CC;
  const unsigned short* bpanel = qkv_wT + (size_t)tn0 * 16 * CC;
  __builtin_prefetch(bpanel, 0, 1);
  f32x8 acc[4] = {zero8(), zero8(), zero8(), zero8()};
#pragma unroll
  for (int kk = 0; kk < CC; kk += 32) {
    bf16x16 a = load_a_frag(abase + kk, CC, lane);
#pragma unroll
    for (int j = 0; j < 4; ++j)
      acc[j] = wmma_bf16(a, load_b_frag(bpanel + (size_t)j * 16 * CC + kk, CC, lane), acc[j]);
  }
  int win = tm / MT_WIN, nr0 = (tm - win * MT_WIN) * 16;
#pragma unroll
  for (int j = 0; j < 4; ++j) {
    int ncol = (tn0 + j) * 16 + (lane & 15);   // 0..383
    int part = ncol >> 7;
    int ccol = ncol & 127;
    int head = ccol >> 4, hd = ccol & 15;
    float bias = qkv_b[ncol];
    size_t base = (size_t)(win * NHD + head);
#pragma unroll
    for (int g = 0; g < 8; ++g) {
      int nrow = nr0 + g + ((lane >> 4) << 3);
      float val = acc[j][g] + bias;
      if (part == 0)      qs[(base * NPAD + nrow) * HDM + hd] = f2bf(val * SCALE);
      else if (part == 1) ks[(base * NPAD + nrow) * HDM + hd] = f2bf(val);
      else                vT[(base * HDM + hd) * NPAD + nrow] = f2bf(val);
    }
  }
}

// ---------------- kernel 3: attention (S = qk^T + bias + mask, softmax, P@V) ----------------
__global__ __launch_bounds__(256) void attn_kernel(const unsigned short* __restrict__ qs,
                                                   const unsigned short* __restrict__ ks,
                                                   const unsigned short* __restrict__ vT,
                                                   const float* __restrict__ rpb,
                                                   const int* __restrict__ rel_index,
                                                   const float* __restrict__ mask,
                                                   unsigned short* __restrict__ obuf) {
  __shared__ float          sS[16 * NPAD];
  __shared__ unsigned short sP[16 * NPAD];
  int blk = blockIdx.x;                    // NWIN * NHD * 22
  int rb = blk % MT_WIN;
  int wh_ = blk / MT_WIN;
  int head = wh_ & 7, win = wh_ >> 3;
  int m0 = rb * 16;
  int wid = threadIdx.x >> 5, lane = threadIdx.x & 31;
  size_t base = (size_t)(win * NHD + head);
  const unsigned short* qbase = qs + base * NPAD * HDM;
  const unsigned short* kbase = ks + base * NPAD * HDM;
  const unsigned short* vbase = vT + base * HDM * NPAD;
  int mw = win & 31;

  // Phase A: one row-tile of S (16 x 352), K = head-dim 16 padded to 32
  bf16x16 afr = load_a_frag_k16(qbase + (size_t)m0 * HDM, HDM, lane);
  for (int tn = wid; tn < MT_WIN; tn += 8) {
    bf16x16 bfr = load_b_frag_k16(kbase + (size_t)tn * 16 * HDM, HDM, lane);
    f32x8 c = zero8();
    c = wmma_bf16(afr, bfr, c);
    int col = tn * 16 + (lane & 15);
#pragma unroll
    for (int g = 0; g < 8; ++g) {
      int row = m0 + g + ((lane >> 4) << 3);
      float v;
      if (row < NT && col < NT) {
        int ri = rel_index[row * NT + col];
        v = c[g] + rpb[ri * NHD + head] + mask[((size_t)mw * NT + row) * NT + col];
      } else {
        v = -1e30f;
      }
      sS[(row - m0) * NPAD + col] = v;
    }
  }
  __syncthreads();

  // Phase B: softmax over columns (16 threads / row)
  {
    int row = threadIdx.x >> 4, sub = threadIdx.x & 15;
    float* sr = sS + row * NPAD;
    float mx = -1e30f;
    for (int ci = sub; ci < NPAD; ci += 16) mx = fmaxf(mx, sr[ci]);
#pragma unroll
    for (int o = 8; o >= 1; o >>= 1) mx = fmaxf(mx, __shfl_xor(mx, o, 32));
    float sum = 0.f;
    for (int ci = sub; ci < NPAD; ci += 16) {
      float e = __expf(sr[ci] - mx);
      sr[ci] = e; sum += e;
    }
#pragma unroll
    for (int o = 8; o >= 1; o >>= 1) sum += __shfl_xor(sum, o, 32);
    float inv = 1.f / sum;
    unsigned short* pr = sP + row * NPAD;
    for (int ci = sub; ci < NPAD; ci += 16) pr[ci] = f2bf(sr[ci] * inv);
  }
  __syncthreads();

  // Phase C: O(16x16) = P(16x352) @ V(352x16), 11 K-steps, A-frags from LDS
  if (wid == 0) {
    f32x8 c = zero8();
#pragma unroll
    for (int kk = 0; kk < NPAD; kk += 32)
      c = wmma_bf16(load_a_frag(sP + kk, NPAD, lane), load_b_frag(vbase + kk, NPAD, lane), c);
    int col = head * HDM + (lane & 15);
#pragma unroll
    for (int g = 0; g < 8; ++g) {
      int row = m0 + g + ((lane >> 4) << 3);
      obuf[((size_t)win * NPAD + row) * CC + col] = f2bf(c[g]);
    }
  }
}

// ---------------- kernel 4: proj GEMM (16x64 strip) + window reverse + residual ----------------
__global__ __launch_bounds__(256) void proj_kernel(const unsigned short* __restrict__ obuf,
                                                   const unsigned short* __restrict__ proj_wT,
                                                   const float* __restrict__ proj_b,
                                                   const float* __restrict__ x,
                                                   float* __restrict__ xo) {
  int wid = threadIdx.x >> 5, lane = threadIdx.x & 31;
  int strip = blockIdx.x * 8 + wid;        // MT_TOT * 2 strips (8 N-tiles / 4)
  if (strip >= MT_TOT * 2) return;
  int tm = strip >> 1, tg = strip & 1;
  int tn0 = tg * 4;
  const unsigned short* abase = obuf + (size_t)tm * 16 * CC;
  const unsigned short* bpanel = proj_wT + (size_t)tn0 * 16 * CC;
  __builtin_prefetch(bpanel, 0, 1);
  f32x8 acc[4] = {zero8(), zero8(), zero8(), zero8()};
#pragma unroll
  for (int kk = 0; kk < CC; kk += 32) {
    bf16x16 a = load_a_frag(abase + kk, CC, lane);
#pragma unroll
    for (int j = 0; j < 4; ++j)
      acc[j] = wmma_bf16(a, load_b_frag(bpanel + (size_t)j * 16 * CC + kk, CC, lane), acc[j]);
  }
  int win = tm / MT_WIN, nr0 = (tm - win * MT_WIN) * 16;
  int b = win >> 5, wr = win & 31;
  int wsI = wr >> 4, whI = (wr >> 2) & 3, wwI = wr & 3;
#pragma unroll
  for (int g = 0; g < 8; ++g) {
    int n = nr0 + g + ((lane >> 4) << 3);
    if (n >= NT) continue;
    int i = n / 49, rem = n - i * 49, j2 = rem / 7, k = rem - j2 * 7;
    int ss = (wsI * WSZ + i + SHF) % SS;
    int hh = (whI * WSZ + j2 + SHF) % HH;
    int wc = (wwI * WSZ + k + SHF) % WWD;
    size_t tk = (size_t)(((b * SS + ss) * HH + hh) * WWD + wc);
#pragma unroll
    for (int j = 0; j < 4; ++j) {
      int col = (tn0 + j) * 16 + (lane & 15);
      xo[tk * CC + col] = x[tk * CC + col] + acc[j][g] + proj_b[col];
    }
  }
}

// ---------------- kernel 5: LN2 -> bf16 ----------------
__global__ __launch_bounds__(256) void ln2_kernel(const float* __restrict__ xo,
                                                  const float* __restrict__ g2,
                                                  const float* __restrict__ b2,
                                                  unsigned short* __restrict__ ln2h) {
  int wid = threadIdx.x >> 5, lane = threadIdx.x & 31;
  int t = blockIdx.x * 8 + wid;
  if (t >= TOK) return;
  const float* row = xo + (size_t)t * CC;
  unsigned short* dst = ln2h + (size_t)t * CC;
  float v[4]; float s = 0.f;
#pragma unroll
  for (int r = 0; r < 4; ++r) { v[r] = row[lane + 32 * r]; s += v[r]; }
#pragma unroll
  for (int o = 16; o >= 1; o >>= 1) s += __shfl_xor(s, o, 32);
  float mu = s * (1.f / 128.f);
  float vs = 0.f;
#pragma unroll
  for (int r = 0; r < 4; ++r) { v[r] -= mu; vs += v[r] * v[r]; }
#pragma unroll
  for (int o = 16; o >= 1; o >>= 1) vs += __shfl_xor(vs, o, 32);
  float rstd = rsqrtf(vs * (1.f / 128.f) + EPSV);
#pragma unroll
  for (int r = 0; r < 4; ++r) {
    int c = lane + 32 * r;
    dst[c] = f2bf(v[r] * rstd * g2[c] + b2[c]);
  }
}

// ---------------- kernel 6: fused MLP (fc1 -> gelu -> fc2 -> += xo), 16x64 strips ----------------
__global__ __launch_bounds__(256) void mlp_kernel(const unsigned short* __restrict__ ln2h,
                                                  const unsigned short* __restrict__ fc1_wT,
                                                  const float* __restrict__ fc1_b,
                                                  const unsigned short* __restrict__ fc2_wT,
                                                  const float* __restrict__ fc2_b,
                                                  float* __restrict__ xo) {
  __shared__ unsigned short hbuf[32 * HID];
  int wid = threadIdx.x >> 5, lane = threadIdx.x & 31;
  int r0 = blockIdx.x * 32;                // 686 blocks * 32 rows
  __builtin_prefetch(fc1_wT, 0, 1);
  __builtin_prefetch(fc2_wT, 0, 1);
  // fc1 + gelu -> hbuf : 2 M-tiles x 8 strips of 4 N-tiles
  for (int t = wid; t < 16; t += 8) {
    int tm = t & 1, tg = t >> 1;
    int tn0 = tg * 4;
    const unsigned short* abase = ln2h + (size_t)(r0 + tm * 16) * CC;
    const unsigned short* bpanel = fc1_wT + (size_t)tn0 * 16 * CC;
    f32x8 acc[4] = {zero8(), zero8(), zero8(), zero8()};
#pragma unroll
    for (int kk = 0; kk < CC; kk += 32) {
      bf16x16 a = load_a_frag(abase + kk, CC, lane);
#pragma unroll
      for (int j = 0; j < 4; ++j)
        acc[j] = wmma_bf16(a, load_b_frag(bpanel + (size_t)j * 16 * CC + kk, CC, lane), acc[j]);
    }
#pragma unroll
    for (int j = 0; j < 4; ++j) {
      int col = (tn0 + j) * 16 + (lane & 15);
      float fb = fc1_b[col];
#pragma unroll
      for (int g = 0; g < 8; ++g) {
        int row = tm * 16 + g + ((lane >> 4) << 3);
        float xv = acc[j][g] + fb;
        float ge = 0.5f * xv * (1.f + erff(xv * 0.70710678118654752f));
        hbuf[row * HID + col] = f2bf(ge);
      }
    }
  }
  __syncthreads();
  // fc2 -> accumulate into xo : 2 M-tiles x 2 strips of 4 N-tiles
  for (int t = wid; t < 4; t += 8) {
    int tm = t & 1, tg = t >> 1;
    int tn0 = tg * 4;
    const unsigned short* abase = hbuf + (size_t)tm * 16 * HID;
    const unsigned short* bpanel = fc2_wT + (size_t)tn0 * 16 * HID;
    f32x8 acc[4] = {zero8(), zero8(), zero8(), zero8()};
#pragma unroll
    for (int kk = 0; kk < HID; kk += 32) {
      bf16x16 a = load_a_frag(abase + kk, HID, lane);
#pragma unroll
      for (int j = 0; j < 4; ++j)
        acc[j] = wmma_bf16(a, load_b_frag(bpanel + (size_t)j * 16 * HID + kk, HID, lane), acc[j]);
    }
#pragma unroll
    for (int j = 0; j < 4; ++j) {
      int col = (tn0 + j) * 16 + (lane & 15);
      float fb = fc2_b[col];
#pragma unroll
      for (int g = 0; g < 8; ++g) {
        int row = r0 + tm * 16 + g + ((lane >> 4) << 3);
        xo[(size_t)row * CC + col] += acc[j][g] + fb;
      }
    }
  }
}

// ---------------- host-side launch ----------------
extern "C" void kernel_launch(void* const* d_in, const int* in_sizes, int n_in,
                              void* d_out, int out_size, void* d_ws, size_t ws_size,
                              hipStream_t stream) {
  (void)in_sizes; (void)n_in; (void)out_size; (void)ws_size;
  const float* x      = (const float*)d_in[0];
  const float* maskm  = (const float*)d_in[1];
  const float* g1     = (const float*)d_in[2];
  const float* b1     = (const float*)d_in[3];
  const float* qkv_w  = (const float*)d_in[4];
  const float* qkv_b  = (const float*)d_in[5];
  const float* rpb    = (const float*)d_in[6];
  const float* proj_w = (const float*)d_in[7];
  const float* proj_b = (const float*)d_in[8];
  const float* g2     = (const float*)d_in[9];
  const float* b2     = (const float*)d_in[10];
  const float* fc1_w  = (const float*)d_in[11];
  const float* fc1_b  = (const float*)d_in[12];
  const float* fc2_w  = (const float*)d_in[13];
  const float* fc2_b  = (const float*)d_in[14];
  const int*   relidx = (const int*)d_in[15];
  float* xo = (float*)d_out;

  // carve workspace (bf16 = unsigned short), 256B aligned
  char* wsp = (char*)d_ws;
  size_t off = 0;
  auto carve = [&](size_t elems) {
    unsigned short* p = (unsigned short*)(wsp + off);
    off = (off + elems * sizeof(unsigned short) + 255) & ~(size_t)255;
    return p;
  };
  unsigned short* hwin    = carve((size_t)NWIN * NPAD * CC);
  unsigned short* qs      = carve((size_t)NWIN * NHD * NPAD * HDM);
  unsigned short* ks      = carve((size_t)NWIN * NHD * NPAD * HDM);
  unsigned short* vT      = carve((size_t)NWIN * NHD * HDM * NPAD);
  unsigned short* obuf    = carve((size_t)NWIN * NPAD * CC);
  unsigned short* ln2h    = carve((size_t)TOK * CC);
  unsigned short* qkv_wT  = carve((size_t)384 * CC);
  unsigned short* proj_wT = carve((size_t)CC * CC);
  unsigned short* fc1_wT  = carve((size_t)HID * CC);
  unsigned short* fc2_wT  = carve((size_t)CC * HID);

  // 0) weights -> bf16 Bt layout
  transpose_bf16_kernel<<<(384 * CC + 255) / 256, 256, 0, stream>>>(qkv_w, qkv_wT, CC, 384);
  transpose_bf16_kernel<<<(CC * CC + 255) / 256, 256, 0, stream>>>(proj_w, proj_wT, CC, CC);
  transpose_bf16_kernel<<<(HID * CC + 255) / 256, 256, 0, stream>>>(fc1_w, fc1_wT, CC, HID);
  transpose_bf16_kernel<<<(CC * HID + 255) / 256, 256, 0, stream>>>(fc2_w, fc2_wT, HID, CC);

  // 1) LN1 + roll + window partition
  ln1_window_kernel<<<(NWIN * NPAD) / 8, 256, 0, stream>>>(x, g1, b1, hwin);

  // 2) QKV (16x64 strips)
  qkv_kernel<<<(MT_TOT * 6) / 8, 256, 0, stream>>>(hwin, qkv_wT, qkv_b, qs, ks, vT);

  // 3) attention
  attn_kernel<<<NWIN * NHD * MT_WIN, 256, 0, stream>>>(qs, ks, vT, rpb, relidx, maskm, obuf);

  // 4) proj + window reverse + residual (16x64 strips)
  proj_kernel<<<(MT_TOT * 2) / 8, 256, 0, stream>>>(obuf, proj_wT, proj_b, x, xo);

  // 5) LN2
  ln2_kernel<<<TOK / 8, 256, 0, stream>>>(xo, g2, b2, ln2h);

  // 6) fused MLP, accumulate into output
  mlp_kernel<<<TOK / 32, 256, 0, stream>>>(ln2h, fc1_wT, fc1_b, fc2_wT, fc2_b, xo);
}
